// NeuronLevelModels_26903675142248
// MI455X (gfx1250) — matrix operand order repaired and checked by reading, not compiled, not run on hardware
//
#include <hip/hip_runtime.h>
#include <hip/hip_bf16.h>

// Problem: b=256, d=2048, m=64, h=128
//   out[b,h] = sum_{d,m} x[b,d,m] * w1[m,h,d] * w2[h,d]  (+ b1/b2 bias terms)
// => single GEMM: C(256x128) = A(256x131072) * B(131072x128)
//    A = x flat (k = d*64+m is exactly x's layout), B[k,h] = w1[m,h,d]*w2[h,d]

#define KTOT   131072      // 2048 * 64
#define MROWS  256
#define NCOLS  128
#define SPLITK 256
#define KCHUNK (KTOT / SPLITK)   // 512
#define TILEK  32
#define KSTEPS (KCHUNK / TILEK)  // 16
#define ROW_U16 40               // LDS row pitch in ushorts (80 bytes, 16B aligned)

typedef __attribute__((ext_vector_type(16))) __bf16 v16bf;
typedef __attribute__((ext_vector_type(8)))  float  v8f;

union Frag16 {
    uint4 u[2];
    v16bf v;
};

// f32 pair -> packed bf16: single packed-convert VALU op.
__device__ __forceinline__ unsigned int bf16pair(float a, float b) {
    unsigned int r;
    asm("v_cvt_pk_bf16_f32 %0, %1, %2" : "=v"(r) : "v"(a), "v"(b));
    return r;
}

// flat->LDS byte offset (low 32 bits of the flat address)
__device__ __forceinline__ unsigned int lds_off(const void* p) {
    return (unsigned int)(unsigned long long)p;
}

// ---------------------------------------------------------------------------
// Kernel 1: pack  Wp[h][k] = bf16( w1[m,h,d] * w2[h,d] ),  k = d*64 + m
// grid (64 d-chunks, 128 h), 256 threads.
// ---------------------------------------------------------------------------
__global__ __launch_bounds__(256) void nlm_pack_kernel(
    const float* __restrict__ w1, const float* __restrict__ w2,
    unsigned short* __restrict__ wp)
{
    const int dc = blockIdx.x;       // 0..63  (32 d's per block)
    const int h  = blockIdx.y;       // 0..127
    const int d0 = dc * 32;
    const int tid = threadIdx.x;

    __shared__ float s_w2[32];
    __shared__ unsigned short t[64 * 32];   // [m][dd] bf16, 64B rows

    if (tid < 32) s_w2[tid] = w2[h * 2048 + d0 + tid];
    __syncthreads();

    {
        const int m   = tid >> 2;            // 0..63
        const int dd0 = (tid & 3) * 8;       // 0,8,16,24
        const float* src = w1 + (size_t)m * (128 * 2048) + (size_t)h * 2048 + d0 + dd0;
        float4 f0 = ((const float4*)src)[0];
        float4 f1 = ((const float4*)src)[1];
        float r[8] = {f0.x, f0.y, f0.z, f0.w, f1.x, f1.y, f1.z, f1.w};
        uint4 pk;
        unsigned int* pu = (unsigned int*)&pk;
#pragma unroll
        for (int i = 0; i < 4; ++i)
            pu[i] = bf16pair(r[2 * i]     * s_w2[dd0 + 2 * i],
                             r[2 * i + 1] * s_w2[dd0 + 2 * i + 1]);
        *(uint4*)&t[m * 32 + dd0] = pk;
    }
    __syncthreads();

    {
        // output span: k = d0*64 + p, p in [0,2048), p = dd*64 + m
        const int p0 = tid * 8;
        const int dd = p0 >> 6;              // constant across the 8 elements
        const int mb = p0 & 63;
        union { unsigned short s[8]; uint4 u; } o;
#pragma unroll
        for (int i = 0; i < 8; ++i) o.s[i] = t[(mb + i) * 32 + dd];
        *(uint4*)&wp[(size_t)h * KTOT + (size_t)d0 * 64 + p0] = o.u;
    }
}

// ---------------------------------------------------------------------------
// Kernel 2: bias[h] = b2[h] + sum_d b1[h,d]*w2[h,d]
// ---------------------------------------------------------------------------
__global__ __launch_bounds__(256) void nlm_bias_kernel(
    const float* __restrict__ b1, const float* __restrict__ w2,
    const float* __restrict__ b2, float* __restrict__ bias)
{
    const int h = blockIdx.x;
    __shared__ float red[256];
    float s = 0.f;
    for (int d = threadIdx.x; d < 2048; d += 256)
        s += b1[h * 2048 + d] * w2[h * 2048 + d];
    red[threadIdx.x] = s;
    __syncthreads();
    for (int off = 128; off > 0; off >>= 1) {
        if ((int)threadIdx.x < off) red[threadIdx.x] += red[threadIdx.x + off];
        __syncthreads();
    }
    if (threadIdx.x == 0) bias[h] = red[0] + b2[h];
}

// ---------------------------------------------------------------------------
// Kernel 3: split-K bf16 WMMA GEMM, double-buffered LDS.
// grid = SPLITK workgroups of 512 threads (16 waves).
// A tile: global f32 -> regs -> v_cvt_pk_bf16_f32 -> ds_store.
// B tile: GLOBAL_LOAD_ASYNC_TO_LDS_B128 (ASYNCcnt), raw bf16 bytes.
// ---------------------------------------------------------------------------
__global__ __launch_bounds__(512) void nlm_gemm_kernel(
    const float* __restrict__ x, const unsigned short* __restrict__ wp,
    float* __restrict__ partial)
{
    __shared__ unsigned short lds_a[2][MROWS * ROW_U16];   // 2 x 20 KB
    __shared__ unsigned short lds_b[2][NCOLS * ROW_U16];   // 2 x 10 KB

    const int tid  = threadIdx.x;
    const int wave = tid >> 5;
    const int lane = tid & 31;
    const int laneN    = lane & 15;
    const int laneHalf = lane >> 4;
    const int m0 = wave * 16;
    const size_t kchunk = (size_t)blockIdx.x * KCHUNK;

    // cooperative-load roles
    const int arow = tid >> 1, apart = tid & 1;            // A: 16 f32 / thread
    const int brow = tid >> 2, bpart = tid & 3;            // B: 8 bf16 / thread
    const float* aptr          = x  + (size_t)arow * KTOT + kchunk + apart * 16;
    const unsigned short* bptr = wp + (size_t)brow * KTOT + kchunk + bpart * 8;
    const unsigned a_dst_off = arow * ROW_U16 + apart * 16;
    const unsigned b_dst_off = brow * ROW_U16 + bpart * 8;

    v8f acc[8] = {};
    float4 ra[4];

#define ISSUE_B(step, buf)                                                      \
    do {                                                                        \
        unsigned _d = lds_off(&lds_b[(buf)][b_dst_off]);                        \
        const unsigned short* _g = bptr + (size_t)(step) * TILEK;               \
        asm volatile("global_load_async_to_lds_b128 %0, %1, off"                \
                     :: "v"(_d), "v"(_g) : "memory");                           \
    } while (0)

#define LOAD_A_REGS(step)                                                       \
    do {                                                                        \
        const float4* _s = (const float4*)(aptr + (size_t)(step) * TILEK);      \
        ra[0] = _s[0]; ra[1] = _s[1]; ra[2] = _s[2]; ra[3] = _s[3];             \
    } while (0)

#define STORE_A(buf)                                                            \
    do {                                                                        \
        uint4 _pa, _pb;                                                         \
        unsigned int* _qa = (unsigned int*)&_pa;                                \
        unsigned int* _qb = (unsigned int*)&_pb;                                \
        _qa[0] = bf16pair(ra[0].x, ra[0].y); _qa[1] = bf16pair(ra[0].z, ra[0].w); \
        _qa[2] = bf16pair(ra[1].x, ra[1].y); _qa[3] = bf16pair(ra[1].z, ra[1].w); \
        _qb[0] = bf16pair(ra[2].x, ra[2].y); _qb[1] = bf16pair(ra[2].z, ra[2].w); \
        _qb[2] = bf16pair(ra[3].x, ra[3].y); _qb[3] = bf16pair(ra[3].z, ra[3].w); \
        uint4* _dst = (uint4*)&lds_a[(buf)][a_dst_off];                         \
        _dst[0] = _pa; _dst[1] = _pb;                                           \
    } while (0)

    // ---- prologue: fill buffer 0 ----
    LOAD_A_REGS(0);
    ISSUE_B(0, 0);
    STORE_A(0);
    asm volatile("s_wait_asynccnt 0x0" ::: "memory");
    __syncthreads();

    int buf = 0;
    for (int step = 0; step < KSTEPS; ++step) {
        // ---- issue next-step loads into buf^1 ----
        if (step + 1 < KSTEPS) {
            LOAD_A_REGS(step + 1);
            ISSUE_B(step + 1, buf ^ 1);
        }

        // ---- compute from buf ----
        // A fragment: 16x32, lane = M%16; k runs laneHalf*8 and 16+laneHalf*8
        Frag16 a;
        {
            const unsigned short* base = &lds_a[buf][(m0 + laneN) * ROW_U16 + laneHalf * 8];
            a.u[0] = *(const uint4*)(base);
            a.u[1] = *(const uint4*)(base + 16);
        }
        // 8 N-tiles: B fragment 32x16, lane = N%16; k runs laneHalf*8 and 16+laneHalf*8.
        // Register-double-buffer fragments so each WMMA's ds_loads are issued
        // one full WMMA ahead of their use.
        const unsigned short* bbase = &lds_b[buf][laneN * ROW_U16 + laneHalf * 8];
        Frag16 bf[2];
        bf[0].u[0] = *(const uint4*)(bbase);
        bf[0].u[1] = *(const uint4*)(bbase + 16);
#pragma unroll
        for (int nt = 0; nt < 8; ++nt) {
            if (nt + 1 < 8) {
                const unsigned short* bb = bbase + (nt + 1) * 16 * ROW_U16;
                bf[(nt + 1) & 1].u[0] = *(const uint4*)(bb);
                bf[(nt + 1) & 1].u[1] = *(const uint4*)(bb + 16);
            }
            acc[nt] = __builtin_amdgcn_wmma_f32_16x16x32_bf16(
                false, a.v, false, bf[nt & 1].v, (short)0, acc[nt], false, false);
        }

        // ---- finish next-step fill, then one barrier per step ----
        if (step + 1 < KSTEPS) STORE_A(buf ^ 1);
        asm volatile("s_wait_asynccnt 0x0" ::: "memory");
        __syncthreads();
        buf ^= 1;
    }

    // ---- store partial C: lanes 0-15 hold M=v, lanes 16-31 hold M=v+8 ----
    float* pp = partial + (size_t)blockIdx.x * (MROWS * NCOLS);
#pragma unroll
    for (int nt = 0; nt < 8; ++nt) {
#pragma unroll
        for (int v = 0; v < 8; ++v) {
            const int M = m0 + v + laneHalf * 8;
            const int N = nt * 16 + laneN;
            pp[M * NCOLS + N] = acc[nt][v];
        }
    }
#undef ISSUE_B
#undef LOAD_A_REGS
#undef STORE_A
}

// ---------------------------------------------------------------------------
// Kernel 4: reduce SPLITK partials + bias -> out
// ---------------------------------------------------------------------------
__global__ __launch_bounds__(256) void nlm_reduce_kernel(
    const float* __restrict__ partial, const float* __restrict__ bias,
    float* __restrict__ out)
{
    const int idx = blockIdx.x * 256 + threadIdx.x;   // 0..32767
    float s = bias[idx & (NCOLS - 1)];
    for (int g = 0; g < SPLITK; ++g)
        s += partial[(size_t)g * (MROWS * NCOLS) + idx];
    out[idx] = s;
}

// ---------------------------------------------------------------------------
extern "C" void kernel_launch(void* const* d_in, const int* in_sizes, int n_in,
                              void* d_out, int out_size, void* d_ws, size_t ws_size,
                              hipStream_t stream) {
    const float* x  = (const float*)d_in[0];
    const float* w1 = (const float*)d_in[1];
    const float* b1 = (const float*)d_in[2];
    const float* w2 = (const float*)d_in[3];
    const float* b2 = (const float*)d_in[4];
    float* out = (float*)d_out;

    // workspace layout
    const size_t wp_bytes      = (size_t)KTOT * NCOLS * sizeof(unsigned short);  // 32 MiB
    const size_t partial_bytes = (size_t)SPLITK * MROWS * NCOLS * sizeof(float); // 32 MiB
    unsigned short* wp = (unsigned short*)d_ws;
    float* partial = (float*)((char*)d_ws + wp_bytes);
    float* bias    = (float*)((char*)d_ws + wp_bytes + partial_bytes);

    nlm_pack_kernel<<<dim3(64, 128), 256, 0, stream>>>(w1, w2, wp);
    nlm_bias_kernel<<<128, 256, 0, stream>>>(b1, w2, b2, bias);
    nlm_gemm_kernel<<<SPLITK, 512, 0, stream>>>(x, wp, partial);
    nlm_reduce_kernel<<<(MROWS * NCOLS) / 256, 256, 0, stream>>>(partial, bias, out);
}
